// Net_49649821942221
// MI455X (gfx1250) — matrix operand order repaired and checked by reading, not compiled, not run on hardware
//
#include <hip/hip_runtime.h>

// Problem dims
#define Bsz 16384
#define Nn  16
#define Hh  512

typedef __attribute__((ext_vector_type(16))) __bf16       bf16x16;
typedef __attribute__((ext_vector_type(8)))  float        f32x8;
typedef __attribute__((ext_vector_type(4)))  unsigned int u32x4;

union Frag32B { u32x4 q[2]; bf16x16 v; };   // 32 bytes: one WMMA bf16 A/B fragment per lane

__device__ __forceinline__ unsigned short f2bf(float f) {
    unsigned int u = __float_as_uint(f);
    u += 0x7FFFu + ((u >> 16) & 1u);        // round-to-nearest-even
    return (unsigned short)(u >> 16);
}

// ---------------------------------------------------------------------------
// K0: convert W2a / W2b (512x512 f32 row-major) to bf16 row-major in ws
// ---------------------------------------------------------------------------
__global__ __launch_bounds__(256) void cvt_w2(const float* __restrict__ A,
                                              const float* __restrict__ Bm,
                                              unsigned short* __restrict__ a,
                                              unsigned short* __restrict__ b) {
    const int i = blockIdx.x * 256 + threadIdx.x;   // 262144 elems
    a[i] = f2bf(A[i]);
    b[i] = f2bf(Bm[i]);
}

// ---------------------------------------------------------------------------
// K1: layer 1 (K=16, trivial FLOPs): z = x@W1^T + b1, t = xdot@W1^T
//     y1 = z*z ; u = 2*z*t  -> bf16 to ws  (A operands of the big GEMMs)
// ---------------------------------------------------------------------------
__global__ __launch_bounds__(256) void layer1_kernel(const float* __restrict__ X,
                                                     const float* __restrict__ Xd,
                                                     const float* __restrict__ W1,
                                                     const float* __restrict__ b1,
                                                     unsigned short* __restrict__ y1o,
                                                     unsigned short* __restrict__ uo) {
    const int row = blockIdx.x >> 1;                         // uniform per block
    const int h   = ((blockIdx.x & 1) << 8) | threadIdx.x;   // 0..511
    const float4* __restrict__ wr = (const float4*)(W1 + h * Nn);
    const float4* __restrict__ xr = (const float4*)(X  + row * Nn);
    const float4* __restrict__ xd = (const float4*)(Xd + row * Nn);
    float z = b1[h], t = 0.f;
#pragma unroll
    for (int q = 0; q < 4; ++q) {
        const float4 w = wr[q], xv = xr[q], dv = xd[q];
        z += w.x * xv.x + w.y * xv.y + w.z * xv.z + w.w * xv.w;
        t += w.x * dv.x + w.y * dv.y + w.z * dv.z + w.w * dv.w;
    }
    const float y1 = z * z;
    const float u  = 2.f * z * t;
    y1o[row * Hh + h] = f2bf(y1);
    uo [row * Hh + h] = f2bf(u);
}

// ---------------------------------------------------------------------------
// K2: fused quad-GEMM on WMMA bf16:
//   z1 = y1@W2a^T+b2a, z2 = y1@W2b^T+b2b, ua = u@W2a^T, ub = u@W2b^T
//   y2 = z1*z2 (stored), v = z1*ub + z2*ua
//   partial row dots with Wout -> ws partials (deterministic, no atomics)
// WG: 64 rows x 128 cols, 8 waves, each wave: 2x2 16x16 tiles x 4 accum sets.
// ---------------------------------------------------------------------------
__global__ __launch_bounds__(256) void fused_gemm(
        const unsigned short* __restrict__ w2a,   // [512][512] bf16 row-major
        const unsigned short* __restrict__ w2b,
        const unsigned short* __restrict__ y1i,   // [B][512] bf16
        const unsigned short* __restrict__ ui,
        const float* __restrict__ b2a, const float* __restrict__ b2b,
        const float* __restrict__ wout,
        float* __restrict__ outY,                 // [B][512] f32
        float* __restrict__ pB, float* __restrict__ pV) {  // [B][16] partials
    extern __shared__ char smem[];
    unsigned short* ldsY = (unsigned short*)smem;     // [64][520] (+8 ushort pad: bank-safe)
    unsigned short* ldsU = ldsY + 64 * 520;

    const int tid     = threadIdx.x;
    const int rowBase = blockIdx.x * 64;
    const int by      = blockIdx.y;                   // 0..3 column blocks of 128

    // Stage A tiles (y1, u) into padded LDS: 64 rows x 512 bf16 each
    for (int i = tid; i < 64 * 64; i += 256) {        // 8-ushort chunks
        const int r = i >> 6;
        const int c = (i & 63) << 3;
        *(u32x4*)(ldsY + r * 520 + c) = *(const u32x4*)(y1i + (rowBase + r) * Hh + c);
        *(u32x4*)(ldsU + r * 520 + c) = *(const u32x4*)(ui  + (rowBase + r) * Hh + c);
    }
    __syncthreads();

    const int lane = tid & 31;
    const int wave = tid >> 5;
    const int wm   = wave >> 2;                // 0..1 : 32-row strip
    const int wn   = wave & 3;                 // 0..3 : 32-col strip
    const int lrow = lane & 15;
    const int hi   = lane >> 4;                // lane half

    const int aRow0 = wm * 32 + lrow;                    // m-tile 0 row (in WG)
    const int col0  = by * 128 + wn * 32 + lrow;         // n-tile 0 h index (global)

    f32x8 acc[2][2][4] = {};                   // [m][n][{y1Wa,y1Wb,uWa,uWb}]

    for (int k0 = 0; k0 < Hh; k0 += 32) {
        Frag32B aF[2][2];                      // [mat(y1,u)][m]
        Frag32B bF[2][2];                      // [mat(Wa,Wb)][n]
        const int ao = k0 + hi * 8;            // ISA A layout: lanes<16 K{0-7,16-23}, >=16 K{8-15,24-31}
        const int bo = k0 + hi * 8;            // B layout matches A: lanes<16 K{0-7,16-23}, >=16 K{8-15,24-31}
#pragma unroll
        for (int m = 0; m < 2; ++m) {
            const int rb = (aRow0 + m * 16) * 520 + ao;
            aF[0][m].q[0] = *(const u32x4*)(ldsY + rb);
            aF[0][m].q[1] = *(const u32x4*)(ldsY + rb + 16);
            aF[1][m].q[0] = *(const u32x4*)(ldsU + rb);
            aF[1][m].q[1] = *(const u32x4*)(ldsU + rb + 16);
        }
#pragma unroll
        for (int n = 0; n < 2; ++n) {
            const int cb = (col0 + n * 16) * Hh + bo;
            bF[0][n].q[0] = *(const u32x4*)(w2a + cb);
            bF[0][n].q[1] = *(const u32x4*)(w2a + cb + 16);
            bF[1][n].q[0] = *(const u32x4*)(w2b + cb);
            bF[1][n].q[1] = *(const u32x4*)(w2b + cb + 16);
        }
#pragma unroll
        for (int m = 0; m < 2; ++m)
#pragma unroll
            for (int n = 0; n < 2; ++n) {
                acc[m][n][0] = __builtin_amdgcn_wmma_f32_16x16x32_bf16(false, aF[0][m].v, false, bF[0][n].v, (short)0, acc[m][n][0], false, false);
                acc[m][n][1] = __builtin_amdgcn_wmma_f32_16x16x32_bf16(false, aF[0][m].v, false, bF[1][n].v, (short)0, acc[m][n][1], false, false);
                acc[m][n][2] = __builtin_amdgcn_wmma_f32_16x16x32_bf16(false, aF[1][m].v, false, bF[0][n].v, (short)0, acc[m][n][2], false, false);
                acc[m][n][3] = __builtin_amdgcn_wmma_f32_16x16x32_bf16(false, aF[1][m].v, false, bF[1][n].v, (short)0, acc[m][n][3], false, false);
            }
    }

    // Epilogue. C/D layout: VGPR r -> row (r + hi*8), col = lrow within tile.
    float wo[2], ba[2], bb[2];
#pragma unroll
    for (int n = 0; n < 2; ++n) {
        const int h = col0 + n * 16;
        wo[n] = wout[h]; ba[n] = b2a[h]; bb[n] = b2b[h];
    }
    const int pcol = by * 4 + wn;
#pragma unroll
    for (int m = 0; m < 2; ++m) {
        const int rb = rowBase + wm * 32 + m * 16 + hi * 8;
#pragma unroll
        for (int r = 0; r < 8; ++r) {
            const int row = rb + r;
            float sb = 0.f, sv = 0.f;
#pragma unroll
            for (int n = 0; n < 2; ++n) {
                const float z1 = acc[m][n][0][r] + ba[n];
                const float z2 = acc[m][n][1][r] + bb[n];
                const float y2 = z1 * z2;
                const float vv = z1 * acc[m][n][3][r] + z2 * acc[m][n][2][r];
                outY[(size_t)row * Hh + (col0 + n * 16)] = y2;
                sb += wo[n] * y2;
                sv += wo[n] * vv;
            }
#pragma unroll
            for (int mk = 1; mk <= 8; mk <<= 1) {   // reduce each 16-lane half
                sb += __shfl_xor(sb, mk, 32);
                sv += __shfl_xor(sv, mk, 32);
            }
            if (lrow == 0) {
                pB[row * 16 + pcol] = sb;
                pV[row * 16 + pcol] = sv;
            }
        }
    }
}

// ---------------------------------------------------------------------------
// K3: sum 16 partials/row -> numerical_b, numerical_bdot
// ---------------------------------------------------------------------------
__global__ __launch_bounds__(256) void reduce_out(const float* __restrict__ pB,
                                                  const float* __restrict__ pV,
                                                  float* __restrict__ out) {
    const int row = blockIdx.x * 256 + threadIdx.x;
    float sb = 0.f, sv = 0.f;
#pragma unroll
    for (int j = 0; j < 16; ++j) { sb += pB[row * 16 + j]; sv += pV[row * 16 + j]; }
    out[row]       = sb;   // numerical_b
    out[Bsz + row] = sv;   // numerical_bdot
}

// ---------------------------------------------------------------------------
extern "C" void kernel_launch(void* const* d_in, const int* in_sizes, int n_in,
                              void* d_out, int out_size, void* d_ws, size_t ws_size,
                              hipStream_t stream) {
    (void)in_sizes; (void)n_in; (void)out_size; (void)ws_size;
    const float* x    = (const float*)d_in[0];
    const float* xdot = (const float*)d_in[1];
    const float* W1a  = (const float*)d_in[2];
    const float* b1a  = (const float*)d_in[3];
    const float* W2a  = (const float*)d_in[4];
    const float* b2a  = (const float*)d_in[5];
    const float* W2b  = (const float*)d_in[6];
    const float* b2b  = (const float*)d_in[7];
    const float* Wout = (const float*)d_in[8];

    char* ws = (char*)d_ws;
    unsigned short* w2a_bf = (unsigned short*)(ws);                      // 512 KB
    unsigned short* w2b_bf = (unsigned short*)(ws + 524288);             // 512 KB
    unsigned short* y1ws   = (unsigned short*)(ws + 1048576);            // 16 MB
    unsigned short* uws    = (unsigned short*)(ws + 1048576 + 16777216); // 16 MB
    float* pB = (float*)(ws + 34603008);                                 // 1 MB
    float* pV = (float*)(ws + 35651584);                                 // 1 MB

    float* out = (float*)d_out;
    // layout: [0,B) numerical_b | [B,2B) bdot | [2B, 2B+B*512) y | then yy=x
    float* outY  = out + 2 * Bsz;
    float* outYY = out + 2 * Bsz + (size_t)Bsz * Hh;

    cvt_w2<<<1024, 256, 0, stream>>>(W2a, W2b, w2a_bf, w2b_bf);
    layer1_kernel<<<Bsz * 2, 256, 0, stream>>>(x, xdot, W1a, b1a, y1ws, uws);

    dim3 g2(Bsz / 64, 4);
    const size_t ldsBytes = 2 * 64 * 520 * sizeof(unsigned short);  // 133,120 B
    fused_gemm<<<g2, 256, ldsBytes, stream>>>(w2a_bf, w2b_bf, y1ws, uws,
                                              b2a, b2b, Wout, outY, pB, pV);

    reduce_out<<<Bsz / 256, 256, 0, stream>>>(pB, pV, out);
    hipMemcpyAsync(outYY, x, (size_t)Bsz * Nn * sizeof(float),
                   hipMemcpyDeviceToDevice, stream);
}
